// Swin3DViTBlock_87428354277547
// MI455X (gfx1250) — matrix operand order repaired and checked by reading, not compiled, not run on hardware
//
#include <hip/hip_runtime.h>
#include <hip/hip_bf16.h>
#include <math.h>

// ---- problem constants ----
#define TOKENS 32768      // B*H*W*D = 2*32*32*16
#define CDIM   384
#define HEADS  12
#define HD     32         // CDIM / HEADS
#define PSTRIDE (6*CDIM)  // 2304 ada params per batch

typedef __bf16 bf16;
typedef bf16  v16bf __attribute__((ext_vector_type(16)));
typedef float v8f   __attribute__((ext_vector_type(8)));
typedef unsigned int u32x4 __attribute__((ext_vector_type(4)));

union Frag {
    v16bf v;
    u32x4 u[2];
    bf16  s[16];
};

__device__ inline v8f wmma_bf16(v16bf a, v16bf b, v8f c) {
    return __builtin_amdgcn_wmma_f32_16x16x32_bf16(false, a, false, b, (short)0, c,
                                                   false, false);
}

// ---------------- fp32 -> bf16 convert ----------------
__global__ __launch_bounds__(256)
void cvt_bf16_k(const float* __restrict__ in, bf16* __restrict__ out, int n) {
    int i = blockIdx.x * 256 + threadIdx.x;
    if (i < n) out[i] = (bf16)in[i];
}

// ---------------- AdaLN-Zero params: silu(noise) @ ada_w^T + ada_b ----------------
__global__ __launch_bounds__(256)
void ada_k(const float* __restrict__ xn, const float* __restrict__ w,
           const float* __restrict__ bias, float* __restrict__ params) {
    __shared__ float s[256];
    int b = blockIdx.x;
    int t = threadIdx.x;
    float v = xn[b * 256 + t];
    s[t] = v / (1.0f + __expf(-v));   // silu
    __syncthreads();
#pragma unroll
    for (int j = 0; j < 9; ++j) {     // 2304 = 9*256 outputs
        int n = t + 256 * j;
        const float* wr = w + (size_t)n * 256;
        float acc = bias[n];
        for (int k = 0; k < 256; ++k) acc += s[k] * wr[k];
        params[(size_t)b * PSTRIDE + n] = acc;
    }
}

// ---------------- LayerNorm + shift/scale modulation -> bf16 ----------------
// one wave32 per token (384 = 12 elems/lane)
__global__ __launch_bounds__(256)
void ln_mod_k(const float* __restrict__ x, const float* __restrict__ params,
              int shift_off, int scale_off, bf16* __restrict__ out) {
    int t    = blockIdx.x * 8 + (threadIdx.x >> 5);
    int lane = threadIdx.x & 31;
    const float* row = x + (size_t)t * CDIM;
    float v[12];
    float s = 0.f;
#pragma unroll
    for (int i = 0; i < 12; ++i) { v[i] = row[lane + 32 * i]; s += v[i]; }
#pragma unroll
    for (int off = 16; off >= 1; off >>= 1) s += __shfl_xor(s, off, 32);
    float mean = s * (1.0f / 384.0f);
    float s2 = 0.f;
#pragma unroll
    for (int i = 0; i < 12; ++i) { float d = v[i] - mean; s2 += d * d; }
#pragma unroll
    for (int off = 16; off >= 1; off >>= 1) s2 += __shfl_xor(s2, off, 32);
    float inv = rsqrtf(s2 * (1.0f / 384.0f) + 1e-5f);
    int bb = t >> 14;   // 16384 tokens per batch
    const float* sh = params + (size_t)bb * PSTRIDE + shift_off;
    const float* sc = params + (size_t)bb * PSTRIDE + scale_off;
    bf16* orow = out + (size_t)t * CDIM;
#pragma unroll
    for (int i = 0; i < 12; ++i) {
        int c = lane + 32 * i;
        float y = (v[i] - mean) * inv;
        y = y * (1.0f + sc[c]) + sh[c];
        orow[c] = (bf16)y;
    }
}

// ---------------- bf16 WMMA GEMM: out[M,N] = A[M,K] @ W[N,K]^T (+ epilogue) ----
// EPI 0: +bias -> bf16   | EPI 1: +bias, exact GELU -> bf16
// EPI 2: resid + gate*( +bias ) -> f32
// block = 256 threads = 8 waves arranged 4(M) x 2(N); block tile 128 x 128;
// wave tile 32(M) x 64(N) = 8 WMMA per 32-deep k-step, B-frags reused across
// the 2 M-subtiles (1.5 fragment loads per WMMA). K unrolled x2, with
// global_prefetch of the streaming A operand one k-step ahead.
template <int EPI>
__global__ __launch_bounds__(256)
void gemm_bf16_k(const bf16* __restrict__ A, const bf16* __restrict__ W,
                 const float* __restrict__ bias,
                 const float* __restrict__ resid, const float* __restrict__ gate,
                 float* __restrict__ outf, bf16* __restrict__ outb,
                 int N, int K) {
    const int lane = threadIdx.x & 31;
    const int wave = threadIdx.x >> 5;
    const int wm   = wave >> 1;        // 0..3
    const int wn   = wave & 1;         // 0..1
    const int g    = lane >> 4;
    const int h16  = lane & 15;
    const int mo   = blockIdx.y * 128 + wm * 32;
    const int no   = blockIdx.x * 128 + wn * 64;

    v8f acc[2][4] = {};
    const bf16* arow0 = A + (size_t)(mo + h16) * K;
    const bf16* arow1 = A + (size_t)(mo + 16 + h16) * K;

    for (int ko = 0; ko < K; ko += 64) {
        // prefetch next k-step of the streaming A operand (speculative; safe OOB)
        __builtin_prefetch((const void*)(arow0 + ko + 64 + 8 * g), 0, 0);
        __builtin_prefetch((const void*)(arow1 + ko + 64 + 8 * g), 0, 0);
#pragma unroll
        for (int kk = 0; kk < 2; ++kk) {
            int k0 = ko + kk * 32;
            Frag a0, a1;   // A-frag 16x32: lane m=h16, halves {8g..+7, 16+8g..+7}
            a0.u[0] = *(const u32x4*)(arow0 + k0 + 8 * g);
            a0.u[1] = *(const u32x4*)(arow0 + k0 + 16 + 8 * g);
            a1.u[0] = *(const u32x4*)(arow1 + k0 + 8 * g);
            a1.u[1] = *(const u32x4*)(arow1 + k0 + 16 + 8 * g);
#pragma unroll
            for (int t = 0; t < 4; ++t) {
                // B-frag 32x16: lane n=h16, halves k = {8g..+7, 16+8g..+7} (K-contig in W[N,K])
                const bf16* wrow = W + (size_t)(no + t * 16 + h16) * K + k0 + 8 * g;
                Frag b;
                b.u[0] = *(const u32x4*)(wrow);
                b.u[1] = *(const u32x4*)(wrow + 16);
                acc[0][t] = wmma_bf16(a0.v, b.v, acc[0][t]);
                acc[1][t] = wmma_bf16(a1.v, b.v, acc[1][t]);
            }
        }
    }

#pragma unroll
    for (int mh = 0; mh < 2; ++mh) {
#pragma unroll
        for (int t = 0; t < 4; ++t) {
            int n = no + t * 16 + h16;
            float bv = bias[n];
#pragma unroll
            for (int r = 0; r < 8; ++r) {
                int row = mo + mh * 16 + r + 8 * g;  // C/D frag rows r+8g
                float v = acc[mh][t][r] + bv;
                if (EPI == 1) v = 0.5f * v * (1.0f + erff(v * 0.70710678118654752f));
                if (EPI == 2) {
                    int b = row >> 14;
                    v = resid[(size_t)row * N + n] + gate[(size_t)b * PSTRIDE + n] * v;
                    outf[(size_t)row * N + n] = v;
                } else {
                    outb[(size_t)row * N + n] = (bf16)v;
                }
            }
        }
    }
}

// ---------------- windowed attention ----------------
__device__ inline int tok_of(int b, int hb, int wb, int db, int lr) {
    int wh = lr >> 4, ww = (lr >> 2) & 3, wd = lr & 3;
    return ((b * 32 + hb * 4 + wh) * 32 + wb * 4 + ww) * 16 + db * 4 + wd;
}

// one block of 128 threads (4 waves) per (window, head); L=64, hd=32
__global__ __launch_bounds__(128)
void win_attn_k(const bf16* __restrict__ qkv, bf16* __restrict__ aout) {
    __shared__ bf16 Vt[32][64];     // V^T: [channel][token-in-window]
    __shared__ bf16 P[4][16][64];   // per-wave softmax probs

    const int head = blockIdx.x % HEADS;
    const int wid  = blockIdx.x / HEADS;
    const int db = wid & 3, wb = (wid >> 2) & 7, hb = (wid >> 5) & 7, b = wid >> 8;

    // stage V^T into LDS (all 128 threads)
    {
        int idx = threadIdx.x;
        int n  = idx >> 1;
        int ch = (idx & 1) * 16;
        const bf16* vr = qkv + (size_t)tok_of(b, hb, wb, db, n) * (3 * CDIM)
                       + 2 * CDIM + head * HD + ch;
        Frag f;
        f.u[0] = *(const u32x4*)(vr);
        f.u[1] = *(const u32x4*)(vr + 8);
#pragma unroll
        for (int i = 0; i < 16; ++i) Vt[ch + i][n] = f.s[i];
    }
    __syncthreads();

    const int lane = threadIdx.x & 31;
    const int wave = threadIdx.x >> 5;
    const int g = lane >> 4, h16 = lane & 15;

    // Q fragment (A 16x32), rows = wave*16 .. +15
    Frag qa;
    {
        int m = wave * 16 + h16;
        const bf16* qr = qkv + (size_t)tok_of(b, hb, wb, db, m) * (3 * CDIM) + head * HD;
        qa.u[0] = *(const u32x4*)(qr + 8 * g);
        qa.u[1] = *(const u32x4*)(qr + 16 + 8 * g);
    }

    // S = Q K^T  (4 tiles of 16x16, K=32 in one WMMA step)
    v8f st[4] = {};
#pragma unroll
    for (int nt = 0; nt < 4; ++nt) {
        int nrow = nt * 16 + h16;
        const bf16* kr = qkv + (size_t)tok_of(b, hb, wb, db, nrow) * (3 * CDIM)
                       + CDIM + head * HD + 8 * g;
        Frag kb;
        kb.u[0] = *(const u32x4*)(kr);
        kb.u[1] = *(const u32x4*)(kr + 16);
        st[nt] = wmma_bf16(qa.v, kb.v, st[nt]);
    }

    // softmax over 64 cols: row r+8g lives on 16 lanes (xor<=8 stays in-group)
    const float scale = 0.17677669529663687f;   // 1/sqrt(32)
#pragma unroll
    for (int r = 0; r < 8; ++r) {
        float s0 = st[0][r] * scale, s1 = st[1][r] * scale;
        float s2 = st[2][r] * scale, s3 = st[3][r] * scale;
        float mx = fmaxf(fmaxf(s0, s1), fmaxf(s2, s3));
#pragma unroll
        for (int off = 8; off >= 1; off >>= 1) mx = fmaxf(mx, __shfl_xor(mx, off, 32));
        float e0 = __expf(s0 - mx), e1 = __expf(s1 - mx);
        float e2 = __expf(s2 - mx), e3 = __expf(s3 - mx);
        float sum = e0 + e1 + e2 + e3;
#pragma unroll
        for (int off = 8; off >= 1; off >>= 1) sum += __shfl_xor(sum, off, 32);
        float inv = 1.0f / sum;
        int mr = r + 8 * g;
        P[wave][mr][ 0 + h16] = (bf16)(e0 * inv);
        P[wave][mr][16 + h16] = (bf16)(e1 * inv);
        P[wave][mr][32 + h16] = (bf16)(e2 * inv);
        P[wave][mr][48 + h16] = (bf16)(e3 * inv);
    }

    // O = P V  (K=64 -> 2 steps; N=32 -> 2 column tiles)
    v8f oacc[2] = {};
#pragma unroll
    for (int s = 0; s < 2; ++s) {
        Frag pa;
        const bf16* pr = &P[wave][h16][0];
        pa.u[0] = *(const u32x4*)(pr + 32 * s + 8 * g);
        pa.u[1] = *(const u32x4*)(pr + 32 * s + 16 + 8 * g);
#pragma unroll
        for (int ct = 0; ct < 2; ++ct) {
            const bf16* vrow = &Vt[ct * 16 + h16][32 * s + 8 * g];
            Frag vb;
            vb.u[0] = *(const u32x4*)(vrow);
            vb.u[1] = *(const u32x4*)(vrow + 16);
            oacc[ct] = wmma_bf16(pa.v, vb.v, oacc[ct]);
        }
    }

    // scatter O back to [token, C] layout
#pragma unroll
    for (int ct = 0; ct < 2; ++ct) {
#pragma unroll
        for (int r = 0; r < 8; ++r) {
            int m = wave * 16 + r + 8 * g;
            int tok = tok_of(b, hb, wb, db, m);
            aout[(size_t)tok * CDIM + head * HD + ct * 16 + h16] = (bf16)oacc[ct][r];
        }
    }
}

// ---------------- host-side orchestration ----------------
extern "C" void kernel_launch(void* const* d_in, const int* in_sizes, int n_in,
                              void* d_out, int out_size, void* d_ws, size_t ws_size,
                              hipStream_t stream) {
    (void)in_sizes; (void)n_in; (void)out_size; (void)ws_size;
    const float* x       = (const float*)d_in[0];
    const float* x_noise = (const float*)d_in[1];
    const float* ada_w   = (const float*)d_in[2];
    const float* ada_b   = (const float*)d_in[3];
    const float* qkv_w   = (const float*)d_in[4];
    const float* qkv_b   = (const float*)d_in[5];
    const float* out_w   = (const float*)d_in[6];
    const float* out_b   = (const float*)d_in[7];
    const float* mlp_w1  = (const float*)d_in[8];
    const float* mlp_b1  = (const float*)d_in[9];
    const float* mlp_w2  = (const float*)d_in[10];
    const float* mlp_b2  = (const float*)d_in[11];

    char* ws = (char*)d_ws;
    size_t off = 0;
    auto alloc = [&](size_t bytes) -> void* {
        void* p = ws + off;
        off = (off + bytes + 255) & ~(size_t)255;
        return p;
    };

    float* params = (float*)alloc((size_t)2 * PSTRIDE * sizeof(float));
    bf16*  wq  = (bf16*)alloc((size_t)3 * CDIM * CDIM * 2);        // [1152,384]
    bf16*  wo  = (bf16*)alloc((size_t)CDIM * CDIM * 2);            // [384,384]
    bf16*  w1  = (bf16*)alloc((size_t)4 * CDIM * CDIM * 2);        // [1536,384]
    bf16*  w2  = (bf16*)alloc((size_t)CDIM * 4 * CDIM * 2);        // [384,1536]
    bf16*  xmod = (bf16*)alloc((size_t)TOKENS * CDIM * 2);
    bf16*  qkvb = (bf16*)alloc((size_t)TOKENS * 3 * CDIM * 2);
    bf16*  attnb = (bf16*)alloc((size_t)TOKENS * CDIM * 2);
    float* x1   = (float*)alloc((size_t)TOKENS * CDIM * sizeof(float));
    bf16*  hbuf = (bf16*)alloc((size_t)TOKENS * 4 * CDIM * 2);

    // weight converts
    cvt_bf16_k<<<(3*CDIM*CDIM + 255)/256, 256, 0, stream>>>(qkv_w, wq, 3*CDIM*CDIM);
    cvt_bf16_k<<<(CDIM*CDIM   + 255)/256, 256, 0, stream>>>(out_w, wo, CDIM*CDIM);
    cvt_bf16_k<<<(4*CDIM*CDIM + 255)/256, 256, 0, stream>>>(mlp_w1, w1, 4*CDIM*CDIM);
    cvt_bf16_k<<<(CDIM*4*CDIM + 255)/256, 256, 0, stream>>>(mlp_w2, w2, CDIM*4*CDIM);

    // AdaLN params
    ada_k<<<2, 256, 0, stream>>>(x_noise, ada_w, ada_b, params);

    // LN + modulate (shift1/scale1) -> bf16
    ln_mod_k<<<TOKENS/8, 256, 0, stream>>>(x, params, 0, CDIM, xmod);

    // qkv = xmod @ qkv_w^T + qkv_b  -> bf16 [T,1152]
    gemm_bf16_k<0><<<dim3(3*CDIM/128, TOKENS/128), 256, 0, stream>>>(
        xmod, wq, qkv_b, nullptr, nullptr, nullptr, qkvb, 3*CDIM, CDIM);

    // windowed attention (512 windows * 12 heads)
    win_attn_k<<<512 * HEADS, 128, 0, stream>>>(qkvb, attnb);

    // x1 = x + gate1 * (attn @ out_w^T + out_b)  -> f32
    gemm_bf16_k<2><<<dim3(CDIM/128, TOKENS/128), 256, 0, stream>>>(
        attnb, wo, out_b, x, params + 2*CDIM, x1, nullptr, CDIM, CDIM);

    // LN + modulate (shift2/scale2)
    ln_mod_k<<<TOKENS/8, 256, 0, stream>>>(x1, params, 3*CDIM, 4*CDIM, xmod);

    // h = gelu(xmod @ mlp_w1^T + mlp_b1) -> bf16 [T,1536]
    gemm_bf16_k<1><<<dim3(4*CDIM/128, TOKENS/128), 256, 0, stream>>>(
        xmod, w1, mlp_b1, nullptr, nullptr, nullptr, hbuf, 4*CDIM, CDIM);

    // out = x1 + gate2 * (h @ mlp_w2^T + mlp_b2) -> f32 (d_out)
    gemm_bf16_k<2><<<dim3(CDIM/128, TOKENS/128), 256, 0, stream>>>(
        hbuf, w2, mlp_b2, x1, params + 5*CDIM, (float*)d_out, nullptr, CDIM, 4*CDIM);
}
